// MultiHeadSelfCrossAttention_31610959298603
// MI455X (gfx1250) — compile-verified, matrix operand order checked by reading, not run on hardware
//
#include <hip/hip_runtime.h>
#include <hip/hip_bf16.h>

typedef __attribute__((ext_vector_type(16))) __bf16 v16bf;
typedef __attribute__((ext_vector_type(8)))  float  v8f;
typedef unsigned short u16;
typedef unsigned int   u32;

constexpr int BATCH = 2;
constexpr int SEQ   = 4096;
constexpr int DM    = 1024;
constexpr int NH    = 16;
constexpr int HD    = 64;
constexpr int BS    = 256;
constexpr int MROWS = BATCH * SEQ;  // 8192
constexpr int QKVD  = 3 * DM;       // 3072

// ---------- helpers ----------
__device__ __forceinline__ u16 f2bf(float f) {
  u32 u = __builtin_bit_cast(u32, f);
  u += 0x7FFFu + ((u >> 16) & 1u);
  return (u16)(u >> 16);
}
__device__ __forceinline__ u32 pack2(float lo, float hi) {
  return (u32)f2bf(lo) | ((u32)f2bf(hi) << 16);
}
// low 32 bits of a generic pointer to LDS == byte offset usable by DS/async ops
__device__ __forceinline__ u32 lds_addr(const void* p) {
  return (u32)(unsigned long long)p;
}

// async global->LDS 16B copy, GVS mode: saddr(u64) + vaddr(i32). Tracked by ASYNCcnt.
#define ASYNC_B128(ldsaddr, voff, sbase)                                  \
  asm volatile("global_load_async_to_lds_b128 %0, %1, %2"                 \
               :: "v"(ldsaddr), "v"(voff), "s"(sbase) : "memory")
#define WAIT_ASYNC_4() asm volatile("s_wait_asynccnt 0x4" ::: "memory")
#define WAIT_ASYNC_0() asm volatile("s_wait_asynccnt 0x0" ::: "memory")

struct bfpair { uint4 a, b; };

// 16x32 bf16 A/B fragment loader (ISA 16-bit A layout: lane=row, two 16B chunks:
// K = h*8..h*8+7 and 16+h*8.., h = lane>>4).
__device__ __forceinline__ v16bf load_frag(const u16* src, int ld) {
  const int lane = threadIdx.x & 31;
  const int row  = lane & 15;
  const int hh   = lane >> 4;
  const u16* p = src + row * ld + hh * 8;
  bfpair t;
  t.a = *(const uint4*)(p);
  t.b = *(const uint4*)(p + 16);
  return __builtin_bit_cast(v16bf, t);
}

// ---------- f32 -> bf16 pack (8 elements / thread) ----------
__global__ void __launch_bounds__(256)
f32_to_bf16_kernel(const float* __restrict__ src, u16* __restrict__ dst, int n8) {
  const int g = blockIdx.x * 256 + threadIdx.x;
  if (g >= n8) return;
  const float4 f0 = *(const float4*)(src + (size_t)g * 8);
  const float4 f1 = *(const float4*)(src + (size_t)g * 8 + 4);
  uint4 u;
  u.x = pack2(f0.x, f0.y); u.y = pack2(f0.z, f0.w);
  u.z = pack2(f1.x, f1.y); u.w = pack2(f1.z, f1.w);
  *(uint4*)(dst + (size_t)g * 8) = u;
}

// ---------- GEMM: C[M,N] f32 = A[M,K] bf16 x B[N,K]^T bf16 ----------
// 128x128 tile / 256 threads (8 waves: 4 M-groups x 2 N-groups, 2x4 WMMA tiles
// per wave). Double-buffered LDS fed by async global->LDS loads (software
// pipeline: tile k+1 streams while tile k is in the matrix units).
template <int N, int K>
__global__ void __launch_bounds__(256)
wmma_gemm_bf16(const u16* __restrict__ A, const u16* __restrict__ B,
               float* __restrict__ C) {
  constexpr int LDT = 40;       // padded LDS row stride (elements)
  constexpr int KT  = K / 32;   // K tiles
  __shared__ __align__(16) u16 As[2][128 * LDT];
  __shared__ __align__(16) u16 Bs[2][128 * LDT];

  const int tid  = threadIdx.x;
  const int lane = tid & 31;
  const int w    = tid >> 5;
  const int wr   = w & 3;
  const int wc   = w >> 2;
  const int mbase = blockIdx.y * 128;
  const int nbase = blockIdx.x * 128;

  // staging coords: thread covers 8-elem chunks at (row0, q) and (row0+64, q)
  const int row0 = tid >> 2;
  const int q    = (tid & 3) * 8;
  const u32 vA0 = (u32)(((size_t)(mbase + row0) * K + q) * 2);
  const u32 vA1 = vA0 + (u32)(64 * K * 2);
  const u32 vB0 = (u32)(((size_t)(nbase + row0) * K + q) * 2);
  const u32 vB1 = vB0 + (u32)(64 * K * 2);
  u32 ldsA0[2], ldsA1[2], ldsB0[2], ldsB1[2];
#pragma unroll
  for (int bb = 0; bb < 2; ++bb) {
    ldsA0[bb] = lds_addr(&As[bb][row0 * LDT + q]);
    ldsA1[bb] = lds_addr(&As[bb][(row0 + 64) * LDT + q]);
    ldsB0[bb] = lds_addr(&Bs[bb][row0 * LDT + q]);
    ldsB1[bb] = lds_addr(&Bs[bb][(row0 + 64) * LDT + q]);
  }

  v8f acc[2][4];
#pragma unroll
  for (int i = 0; i < 2; ++i)
#pragma unroll
    for (int j = 0; j < 4; ++j) acc[i][j] = {};

  // prologue: stream tile 0 into buffer 0
  ASYNC_B128(ldsA0[0], vA0, A);
  ASYNC_B128(ldsA1[0], vA1, A);
  ASYNC_B128(ldsB0[0], vB0, B);
  ASYNC_B128(ldsB1[0], vB1, B);

  for (int kt = 0; kt < KT; ++kt) {
    const int cur = kt & 1, nxt = cur ^ 1;
    if (kt + 1 < KT) {
      const u16* sa = A + (size_t)(kt + 1) * 32;   // saddr carries k0
      const u16* sb = B + (size_t)(kt + 1) * 32;
      ASYNC_B128(ldsA0[nxt], vA0, sa);
      ASYNC_B128(ldsA1[nxt], vA1, sa);
      ASYNC_B128(ldsB0[nxt], vB0, sb);
      ASYNC_B128(ldsB1[nxt], vB1, sb);
      WAIT_ASYNC_4();   // 4 new in flight; tile `cur` (older, in-order) done
    } else {
      WAIT_ASYNC_0();
    }
    __syncthreads();    // cross-wave visibility of buffer `cur`

    v16bf af[2], bfr[4];
#pragma unroll
    for (int i = 0; i < 2; ++i)
      af[i] = load_frag(&As[cur][(wr * 32 + i * 16) * LDT], LDT);
#pragma unroll
    for (int j = 0; j < 4; ++j)
      bfr[j] = load_frag(&Bs[cur][(wc * 64 + j * 16) * LDT], LDT);
#pragma unroll
    for (int i = 0; i < 2; ++i)
#pragma unroll
      for (int j = 0; j < 4; ++j)
        acc[i][j] = __builtin_amdgcn_wmma_f32_16x16x32_bf16(
            false, af[i], false, bfr[j], (short)0, acc[i][j], false, false);
    __syncthreads();    // buffer `cur` free for the kt+2 async issue
  }

  // epilogue: N is constexpr -> e*N folds into store offset immediates
  const int hl = lane >> 4, nl = lane & 15;
#pragma unroll
  for (int i = 0; i < 2; ++i)
#pragma unroll
    for (int j = 0; j < 4; ++j) {
      float* cp = C + (size_t)(mbase + wr * 32 + i * 16 + 8 * hl) * N +
                  nbase + wc * 64 + j * 16 + nl;
#pragma unroll
      for (int e = 0; e < 8; ++e) cp[e * N] = acc[i][j][e];
    }
}

// ---------- RoPE + head split: qkv f32 -> per-head bf16 q/k/v ----------
__global__ void __launch_bounds__(256)
rope_split_kernel(const float* __restrict__ qkv, u16* __restrict__ qb,
                  u16* __restrict__ kb, u16* __restrict__ vb) {
  int tid = blockIdx.x * 256 + threadIdx.x;
  const int j = tid & 31; tid >>= 5;
  const int h = tid & 15; tid >>= 4;
  const int l = tid & (SEQ - 1);
  const int b = tid >> 12;
  if (b >= BATCH) return;

  const float* row = qkv + (size_t)(b * SEQ + l) * QKVD;
  const float inv_freq = powf(10000.0f, -(float)(2 * j) / 64.0f);
  float s, c;
  __sincosf((float)l * inv_freq, &s, &c);

  const size_t obase = ((size_t)(b * NH + h) * SEQ + l) * HD;

  const float qx1 = row[h * HD + 2 * j], qx2 = row[h * HD + 2 * j + 1];
  qb[obase + j]      = f2bf(qx1 * c - qx2 * s);
  qb[obase + 32 + j] = f2bf(qx2 * c + qx1 * s);

  const float kx1 = row[DM + h * HD + 2 * j], kx2 = row[DM + h * HD + 2 * j + 1];
  kb[obase + j]      = f2bf(kx1 * c - kx2 * s);
  kb[obase + 32 + j] = f2bf(kx2 * c + kx1 * s);

  vb[obase + 2 * j]     = f2bf(row[2 * DM + h * HD + 2 * j]);
  vb[obase + 2 * j + 1] = f2bf(row[2 * DM + h * HD + 2 * j + 1]);
}

// ---------- Block-causal attention, one WG per (b,h,block,64-row q chunk) ----
__global__ void __launch_bounds__(256)
block_attn_kernel(const u16* __restrict__ qb, const u16* __restrict__ kb,
                  const u16* __restrict__ vb, u16* __restrict__ attn) {
  constexpr int LDQ = 72;    // Q/K LDS row stride
  constexpr int LDV = 264;   // V^T / S / P row stride
  extern __shared__ __align__(16) unsigned char smem[];
  u16*   Qs  = (u16*)smem;                  //  64 x 72 bf16
  u16*   Ks  = Qs + 64 * LDQ;               // 256 x 72 bf16
  u16*   VTs = Ks + 256 * LDQ;              //  64 x 264 bf16 (V transposed)
  float* Ss  = (float*)(VTs + 64 * LDV);    //  64 x 264 f32 scores
  u16*   Ps  = Qs;  // P bf16 reuses dead Q+K region after barrier

  const int tid = threadIdx.x, lane = tid & 31, w = tid >> 5;
  int wg = blockIdx.x;
  const int qc  = wg & 3;  wg >>= 2;
  const int blk = wg & 15; wg >>= 4;
  const int h   = wg & 15; wg >>= 4;
  const int b   = wg;
  const size_t headbase = (size_t)(b * NH + h) * SEQ * HD;
  const int l0 = blk * BS;
  const int q0 = qc * 64;

  // --- async-stream Q (64x64) and K (256x64) straight into LDS ---
#pragma unroll
  for (int it = 0; it < 2; ++it) {
    const int g = tid + it * 256, r = g >> 3, col = (g & 7) * 8;
    const u32 voff = (u32)((headbase + (size_t)(l0 + q0 + r) * HD + col) * 2);
    ASYNC_B128(lds_addr(&Qs[r * LDQ + col]), voff, qb);
  }
#pragma unroll
  for (int it = 0; it < 8; ++it) {
    const int g = tid + it * 256, r = g >> 3, col = (g & 7) * 8;
    const u32 voff = (u32)((headbase + (size_t)(l0 + r) * HD + col) * 2);
    ASYNC_B128(lds_addr(&Ks[r * LDQ + col]), voff, kb);
  }
  // --- V: sync load + transposed LDS store (async path can't transpose) ---
#pragma unroll
  for (int it = 0; it < 8; ++it) {
    const int g = tid + it * 256, r = g >> 3, col = (g & 7) * 8;
    uint4 u = *(const uint4*)&vb[headbase + (size_t)(l0 + r) * HD + col];
    const u16* hp = (const u16*)&u;
#pragma unroll
    for (int e = 0; e < 8; ++e) VTs[(col + e) * LDV + r] = hp[e];
  }
  WAIT_ASYNC_0();
  __syncthreads();

  // --- S = Q @ K^T : waves = 4 M-groups(16) x 2 N-groups(128) ---
  const int wr = w & 3, wc = w >> 2;
  v8f sacc[8];
#pragma unroll
  for (int j = 0; j < 8; ++j) sacc[j] = {};
#pragma unroll
  for (int kk = 0; kk < HD; kk += 32) {
    v16bf aq = load_frag(&Qs[(wr * 16) * LDQ + kk], LDQ);
#pragma unroll
    for (int j = 0; j < 8; ++j) {
      v16bf bk = load_frag(&Ks[(wc * 128 + j * 16) * LDQ + kk], LDQ);
      sacc[j] = __builtin_amdgcn_wmma_f32_16x16x32_bf16(
          false, aq, false, bk, (short)0, sacc[j], false, false);
    }
  }
  const int hl = lane >> 4, nl = lane & 15;
  const float scale = 0.125f;  // 64^-0.5
#pragma unroll
  for (int j = 0; j < 8; ++j) {
    const int ncol = wc * 128 + j * 16 + nl;
#pragma unroll
    for (int e = 0; e < 8; ++e) {
      const int mrow = wr * 16 + e + 8 * hl;
      float v = sacc[j][e] * scale;
      if (ncol > q0 + mrow) v = -3.0e38f;
      Ss[mrow * LDV + ncol] = v;
    }
  }
  __syncthreads();

  // --- softmax: wave owns 8 rows; lane handles cols lane+32i ---
#pragma unroll
  for (int rr = 0; rr < 8; ++rr) {
    const int row = w * 8 + rr;
    float vals[8];
    float mx = -3.4e38f;
#pragma unroll
    for (int i = 0; i < 8; ++i) {
      vals[i] = Ss[row * LDV + lane + 32 * i];
      mx = fmaxf(mx, vals[i]);
    }
#pragma unroll
    for (int off = 16; off > 0; off >>= 1) mx = fmaxf(mx, __shfl_xor(mx, off, 32));
    float sum = 0.f;
#pragma unroll
    for (int i = 0; i < 8; ++i) { vals[i] = __expf(vals[i] - mx); sum += vals[i]; }
#pragma unroll
    for (int off = 16; off > 0; off >>= 1) sum += __shfl_xor(sum, off, 32);
    const float inv = 1.0f / sum;
#pragma unroll
    for (int i = 0; i < 8; ++i)
      Ps[row * LDV + lane + 32 * i] = f2bf(vals[i] * inv);
  }
  __syncthreads();

  // --- O = P @ V : 64x64, waves = 4 M-groups x 2 N-groups(32) ---
  v8f oacc[2];
  oacc[0] = {}; oacc[1] = {};
#pragma unroll
  for (int kk = 0; kk < BS; kk += 32) {
    v16bf ap = load_frag(&Ps[(wr * 16) * LDV + kk], LDV);
#pragma unroll
    for (int t = 0; t < 2; ++t) {
      v16bf bv = load_frag(&VTs[(wc * 32 + t * 16) * LDV + kk], LDV);
      oacc[t] = __builtin_amdgcn_wmma_f32_16x16x32_bf16(
          false, ap, false, bv, (short)0, oacc[t], false, false);
    }
  }
#pragma unroll
  for (int t = 0; t < 2; ++t) {
    const int d = wc * 32 + t * 16 + nl;
#pragma unroll
    for (int e = 0; e < 8; ++e) {
      const int m = wr * 16 + e + 8 * hl;
      const int l = l0 + q0 + m;
      attn[(size_t)(b * SEQ + l) * DM + h * HD + d] = f2bf(oacc[t][e]);
    }
  }
}

// ---------- host launch ----------
extern "C" void kernel_launch(void* const* d_in, const int* in_sizes, int n_in,
                              void* d_out, int out_size, void* d_ws, size_t ws_size,
                              hipStream_t stream) {
  const float* x    = (const float*)d_in[0];
  const float* Wqkv = (const float*)d_in[1];
  const float* Wout = (const float*)d_in[2];
  float* out = (float*)d_out;

  char* ws = (char*)d_ws;
  size_t off = 0;
  float* qkv  = (float*)ws;                         // 8192x3072 f32
  u16* attnb  = (u16*)ws;                           // overlays dead qkv
  off += (size_t)MROWS * QKVD * sizeof(float);
  u16* qb = (u16*)(ws + off); off += (size_t)BATCH * NH * SEQ * HD * 2;
  u16* kb = (u16*)(ws + off); off += (size_t)BATCH * NH * SEQ * HD * 2;
  u16* vb = (u16*)(ws + off); off += (size_t)BATCH * NH * SEQ * HD * 2;
  u16* xb    = (u16*)(ws + off); off += (size_t)MROWS * DM * 2;
  u16* wqkvb = (u16*)(ws + off); off += (size_t)QKVD * DM * 2;
  u16* woutb = (u16*)(ws + off); off += (size_t)DM * DM * 2;

  // 0) one-pass bf16 packs
  f32_to_bf16_kernel<<<(MROWS * DM / 8 + 255) / 256, 256, 0, stream>>>(x, xb, MROWS * DM / 8);
  f32_to_bf16_kernel<<<(QKVD * DM / 8 + 255) / 256, 256, 0, stream>>>(Wqkv, wqkvb, QKVD * DM / 8);
  f32_to_bf16_kernel<<<(DM * DM / 8 + 255) / 256, 256, 0, stream>>>(Wout, woutb, DM * DM / 8);

  // 1) qkv = x @ Wqkv^T
  dim3 g1(QKVD / 128, MROWS / 128);
  wmma_gemm_bf16<QKVD, DM><<<g1, 256, 0, stream>>>(xb, wqkvb, qkv);

  // 2) RoPE + split heads to bf16
  rope_split_kernel<<<(BATCH * SEQ * NH * 32) / 256, 256, 0, stream>>>(qkv, qb, kb, vb);

  // 3) block-causal attention (writes bf16, overlays dead qkv)
  const int attn_lds = (64 * 72 + 256 * 72 + 64 * 264) * 2 + 64 * 264 * 4;  // 147456
  hipFuncSetAttribute((const void*)block_attn_kernel,
                      hipFuncAttributeMaxDynamicSharedMemorySize, attn_lds);
  block_attn_kernel<<<BATCH * NH * (SEQ / BS) * (BS / 64), 256, attn_lds, stream>>>(
      qb, kb, vb, attnb);

  // 4) out = attn @ Wout^T
  dim3 g2(DM / 128, MROWS / 128);
  wmma_gemm_bf16<DM, DM><<<g2, 256, 0, stream>>>(attnb, woutb, out);
}
